// Encoder_7713761263714
// MI455X (gfx1250) — compile-verified, matrix operand order checked by reading, not run on hardware
//
#include <hip/hip_runtime.h>
#include <hip/hip_bf16.h>
#include <stdint.h>

// ---------------------------------------------------------------------------
// MI455X (gfx1250) VQ-VAE forward.
//  * Conv layers -> implicit-GEMM, 4-wave blocks, 128x16 tile, double-buffered
//    LDS, 2x v_wmma_f32_16x16x32_f16 per wave per K-step.
//  * All H/W/C are powers of two -> shift/mask addressing, no integer division
//    in any hot loop (round-2 disasm showed rcp-based div chains; eliminated).
//  * Per-thread A-row (m) decode hoisted out of the K loop; K-tap decode is a
//    carried increment. LDS staging via b128 stores/loads.
//  * VQ codebooks: E0 via TDM (tensor_load_to_lds, TENSORcnt),
//    E1/E2 via global_load_async_to_lds_b128 (ASYNCcnt).
// ---------------------------------------------------------------------------

typedef __attribute__((ext_vector_type(16))) _Float16 v16h;
typedef __attribute__((ext_vector_type(8)))  _Float16 v8h;
typedef __attribute__((ext_vector_type(8)))  float    v8f;
typedef __attribute__((ext_vector_type(4)))  unsigned v4u;
typedef __attribute__((ext_vector_type(8)))  unsigned v8u;

#define F_RELU 1
#define F_ADD  2

// ------------------------- generic WMMA conv -------------------------------
// Block = 128 threads (4 waves). Tile = 128(M: spatial) x 16(N: out-channel).
// hwsh = log2(Hout*Wout), wsh = log2(Wout). K = Cin*KH*KW, 32 per step.
__global__ __launch_bounds__(128)
void conv_wmma(const float* __restrict__ x, const float* __restrict__ w,
               const float* __restrict__ bias, float* __restrict__ out,
               int N, int Cin, int Hin, int Win,
               int Cout, int hwsh, int wsh,
               int KH, int KW, int stride, int pad,
               int co_off, int Ctot, int flags)
{
    __shared__ _Float16 As[2][128][32];   // [buf][m_local][k_local]
    __shared__ _Float16 Bs[2][16][32];    // [buf][n_local][k_local]

    const int tid  = threadIdx.x;
    const int lane = tid & 31;
    const int wave = tid >> 5;
    const int tm = blockIdx.x, tn = blockIdx.y;
    const int HW = 1 << hwsh;
    const int M  = N << hwsh;
    const int K  = Cin * KH * KW;
    const int Ksteps = (K + 31) >> 5;

    // ---- hoisted per-thread A-row decode (one shift/mask decode, no div) --
    const int m   = tm * 128 + tid;
    const bool mok = m < M;
    const int n_  = m >> hwsh;
    const int r2_ = m & (HW - 1);
    const int oy  = r2_ >> wsh;
    const int ox  = r2_ & ((1 << wsh) - 1);
    const int iy0 = oy * stride - pad;
    const int ix0 = ox * stride - pad;
    const float* xrow = x + (size_t)n_ * Cin * Hin * Win;

    // carried K-tap state (incremented, never divided)
    int cur_k = 0, cur_ci = 0, cur_ky = 0, cur_kx = 0;

    auto fill = [&](int kt, int buf) {
        // ---- A: this thread owns row `tid`, fills all 32 k columns
        v8h t4[4];
        #pragma unroll
        for (int j = 0; j < 32; ++j) {
            float av = 0.f;
            if (mok && cur_k < K) {
                const int iy = iy0 + cur_ky;
                const int ix = ix0 + cur_kx;
                if ((unsigned)iy < (unsigned)Hin && (unsigned)ix < (unsigned)Win)
                    av = xrow[((size_t)cur_ci * Hin + iy) * Win + ix];
            }
            t4[j >> 3][j & 7] = (_Float16)av;
            ++cur_k;
            if (++cur_kx == KW) { cur_kx = 0; if (++cur_ky == KH) { cur_ky = 0; ++cur_ci; } }
        }
        #pragma unroll
        for (int q = 0; q < 4; ++q)
            *(v8h*)&As[buf][tid][q * 8] = t4[q];        // ds_store_b128

        // ---- B: threads 0..63 each store 8 contiguous weights (b128)
        if (tid < 64) {
            const int nrow = tid >> 2, seg = tid & 3;
            const int co = tn * 16 + nrow;
            const int kb = (kt << 5) + seg * 8;
            v8h bv;
            #pragma unroll
            for (int j = 0; j < 8; ++j) {
                const int k = kb + j;
                bv[j] = (_Float16)((co < Cout && k < K) ? w[(size_t)co * K + k] : 0.f);
            }
            *(v8h*)&Bs[buf][nrow][seg * 8] = bv;
        }
    };

    v8f acc0 = {}, acc1 = {};
    fill(0, 0);
    for (int kt = 0; kt < Ksteps; ++kt) {
        __syncthreads();
        const int buf = kt & 1;
        if (kt + 1 < Ksteps) {
            __builtin_prefetch((const char*)(w + (size_t)(tn * 16) * K + (kt + 1) * 32), 0, 0);
            fill(kt + 1, buf ^ 1);                 // overlap staging with WMMA
        }
        // fragment loads per CDNA5 16-bit A/B layout:
        // lane<16 : row=lane,    K = {0..7, 16..23}
        // lane>=16: row=lane-16, K = {8..15, 24..31}
        const int hh = lane >> 4, r = lane & 15;
        const int r0 = wave * 32 + r, r1 = r0 + 16;
        v8h blo = *(const v8h*)&Bs[buf][r][hh << 3];
        v8h bhi = *(const v8h*)&Bs[buf][r][16 + (hh << 3)];
        v8h a0lo = *(const v8h*)&As[buf][r0][hh << 3];
        v8h a0hi = *(const v8h*)&As[buf][r0][16 + (hh << 3)];
        v8h a1lo = *(const v8h*)&As[buf][r1][hh << 3];
        v8h a1hi = *(const v8h*)&As[buf][r1][16 + (hh << 3)];
        v16h b  = __builtin_shufflevector(blo,  bhi,  0,1,2,3,4,5,6,7,8,9,10,11,12,13,14,15);
        v16h a0 = __builtin_shufflevector(a0lo, a0hi, 0,1,2,3,4,5,6,7,8,9,10,11,12,13,14,15);
        v16h a1 = __builtin_shufflevector(a1lo, a1hi, 0,1,2,3,4,5,6,7,8,9,10,11,12,13,14,15);
        acc0 = __builtin_amdgcn_wmma_f32_16x16x32_f16(false, a0, false, b,
                                                      (short)0, acc0, false, false);
        acc1 = __builtin_amdgcn_wmma_f32_16x16x32_f16(false, a1, false, b,
                                                      (short)0, acc1, false, false);
    }

    // ---- epilogue: C/D layout — lane L: N = L&15, VGPR v: M = v + 8*(L>>4)
    const int hh  = lane >> 4;
    const int col = lane & 15;
    const int co  = tn * 16 + col;
    if (co < Cout) {
        const float bb = bias[co];
        #pragma unroll
        for (int half = 0; half < 2; ++half) {
            const v8f& acc = half ? acc1 : acc0;
            #pragma unroll
            for (int v = 0; v < 8; ++v) {
                const int mm = tm * 128 + wave * 32 + half * 16 + v + (hh << 3);
                if (mm < M) {
                    const int nn = mm >> hwsh;
                    const int rr = mm & (HW - 1);
                    size_t addr = ((size_t)(nn * Ctot + co_off + co) << hwsh) + rr;
                    float val = acc[v] + bb;
                    if (flags & F_ADD)  val += out[addr];
                    if (flags & F_RELU) val = fmaxf(val, 0.f);
                    out[addr] = val;
                }
            }
        }
    }
}

// -------- fold 1x1 s2 skip into tap (1,1) of a 4x4 s2 SAME conv ------------
__global__ void fold_rc(const float* __restrict__ w,  const float* __restrict__ b,
                        const float* __restrict__ wsk, const float* __restrict__ bsk,
                        float* __restrict__ wf, float* __restrict__ bf,
                        int Cout, int cish)            // cish = log2(Cin)
{
    const int i = blockIdx.x * blockDim.x + threadIdx.x;
    const int tot = Cout << (cish + 4);
    if (i < tot) {
        const int t  = i & 15;
        const int ci = (i >> 4) & ((1 << cish) - 1);
        const int co = i >> (4 + cish);
        float v = w[i];
        if (t == 5) v += wsk[(co << cish) + ci];   // tap (ky=1,kx=1) == skip sample
        wf[i] = v;
    }
    if (i < Cout) bf[i] = b[i] + bsk[i];
}

// ----------------- fused res0 (1 -> 32 -> 4, 3x3, 256x256) -----------------
__global__ __launch_bounds__(256)
void res0_fused(const float* __restrict__ x,
                const float* __restrict__ w1, const float* __restrict__ b1,
                const float* __restrict__ w2, const float* __restrict__ b2,
                const float* __restrict__ wsk, const float* __restrict__ bsk,
                float* __restrict__ out, int H, int W)
{
    __shared__ float xs[20][20];
    __shared__ float hs[18][18][32];           // ~43 KB of the 320 KB WGP LDS
    const int n   = blockIdx.z;
    const int ty0 = blockIdx.y * 16, tx0 = blockIdx.x * 16;
    const int tid = threadIdx.x;
    const float* xn = x + (size_t)n * H * W;

    for (int i = tid; i < 400; i += 256) {
        int yy = i / 20, xx = i % 20;          // constant divisors -> mul/shift
        int gy = ty0 - 2 + yy, gx = tx0 - 2 + xx;
        xs[yy][xx] = ((unsigned)gy < (unsigned)H && (unsigned)gx < (unsigned)W)
                   ? xn[gy * W + gx] : 0.f;
    }
    __syncthreads();

    for (int i = tid; i < 18 * 18 * 32; i += 256) {
        const int c = i & 31, p = i >> 5, hy = p / 18, hx = p % 18;
        float a = b1[c];
        #pragma unroll
        for (int ky = 0; ky < 3; ++ky)
            #pragma unroll
            for (int kx = 0; kx < 3; ++kx)
                a += xs[hy + ky][hx + kx] * w1[c * 9 + ky * 3 + kx];
        hs[hy][hx][c] = fmaxf(a, 0.f);
    }
    __syncthreads();

    for (int i = tid; i < 16 * 16 * 4; i += 256) {
        const int c = i & 3, p = i >> 2, oy = p >> 4, ox = p & 15;
        float a = b2[c] + bsk[c] + wsk[c] * xs[oy + 2][ox + 2];
        for (int c2 = 0; c2 < 32; ++c2) {
            #pragma unroll
            for (int ky = 0; ky < 3; ++ky)
                #pragma unroll
                for (int kx = 0; kx < 3; ++kx)
                    a += hs[oy + ky][ox + kx][c2] * w2[((c * 32 + c2) * 3 + ky) * 3 + kx];
        }
        out[(((size_t)n * 4 + c) * H + oy + ty0) * W + ox + tx0] = fmaxf(a, 0.f);
    }
}

// ---------- transposed conv k4 s2 + NN-upsampled 1x1 skip + relu -----------
// wsh=log2(Win), hsh=log2(Hin), csh=log2(Cout). Output is 2*Hin x 2*Win.
__global__ __launch_bounds__(256)
void res_trans_k(const float* __restrict__ x, const float* __restrict__ wt,
                 const float* __restrict__ bt, const float* __restrict__ wsk,
                 const float* __restrict__ bsk, float* __restrict__ out,
                 int Nb, int Cin, int wsh, int hsh, int csh,
                 int co_off, int Ctot)
{
    const int Hin = 1 << hsh, Win = 1 << wsh;
    const int Cout = 1 << csh;
    const int owsh = wsh + 1, ohsh = hsh + 1;
    const int Wout = 1 << owsh, Hout = 1 << ohsh;
    const int total = (Nb << csh) << (owsh + ohsh);
    int idx = blockIdx.x * blockDim.x + threadIdx.x;
    if (idx >= total) return;
    const int ox = idx & (Wout - 1);
    const int oy = (idx >> owsh) & (Hout - 1);
    const int co = (idx >> (owsh + ohsh)) & (Cout - 1);
    const int n  = idx >> (owsh + ohsh + csh);

    float a = bt[co];
    for (int ky = 0; ky < 4; ++ky) {
        const int iy2 = oy + 1 - ky;
        if (iy2 & 1) continue;
        const int iy = iy2 >> 1;
        if ((unsigned)iy >= (unsigned)Hin) continue;
        for (int kx = 0; kx < 4; ++kx) {
            const int ix2 = ox + 1 - kx;
            if (ix2 & 1) continue;
            const int ix = ix2 >> 1;
            if ((unsigned)ix >= (unsigned)Win) continue;
            for (int ci = 0; ci < Cin; ++ci)
                a += x[(((size_t)n * Cin + ci) << (hsh + wsh)) + (iy << wsh) + ix]
                   * wt[((ci * Cout + co) * 4 + ky) * 4 + kx];
        }
    }
    float s = bsk[co];
    const int sy = oy >> 1, sx = ox >> 1;
    for (int ci = 0; ci < Cin; ++ci)
        s += wsk[co * Cin + ci] * x[(((size_t)n * Cin + ci) << (hsh + wsh)) + (sy << wsh) + sx];

    out[(((size_t)n * Ctot + co_off + co) << (ohsh + owsh)) + (oy << owsh) + ox]
        = fmaxf(a + s, 0.f);
}

// ----------------------- channel-slice copy (concat) -----------------------
__global__ void copy_ch(const float* __restrict__ src, float* __restrict__ dst,
                        int Nb, int hwsh, int csh, int dst_off, int Ctot)
{
    const int total = Nb << (csh + hwsh);
    int idx = blockIdx.x * blockDim.x + threadIdx.x;
    if (idx >= total) return;
    const int p = idx & ((1 << hwsh) - 1);
    const int c = (idx >> hwsh) & ((1 << csh) - 1);
    const int n = idx >> (hwsh + csh);
    dst[(((size_t)n * Ctot + dst_off + c) << hwsh) + p] = src[idx];
}

// -------------------- TDM: 1-D tile global -> LDS --------------------------
// Builds a Tensor DMA Descriptor (D#) in SGPRs and issues TENSOR_LOAD_TO_LDS.
// group0: count=1 | lds_addr | global_addr[56:0] | type=2 ("image")
// group1: data_size=4B, tensor_dim0 = tile_dim0 = nelems, dim1 = 1.
__device__ inline void tdm_load_1d(const void* src, unsigned lds_off, unsigned nelems)
{
    const uint64_t ga = (uint64_t)(uintptr_t)src;
    v4u g0;
    g0.x = 1u;                                            // count=1, user mode
    g0.y = lds_off;                                       // lds_addr (bytes)
    g0.z = (unsigned)ga;                                  // global_addr[31:0]
    g0.w = (unsigned)((ga >> 32) & 0x01ffffffu) | (2u << 30); // addr[56:32]|type=2
    v8u g1;
    g1[0] = 2u << 16;                                     // data_size = 4 bytes
    g1[1] = (nelems & 0xffffu) << 16;                     // tensor_dim0[15:0]
    g1[2] = (nelems >> 16) | (1u << 16);                  // dim0[31:16] | tensor_dim1=1
    g1[3] = (nelems & 0xffffu) << 16;                     // tile_dim0
    g1[4] = 1u;                                           // tile_dim1=1, tile_dim2=0
    g1[5] = nelems;                                       // tensor_dim0_stride[31:0]
    g1[6] = 0u;
    g1[7] = 0u;
    asm volatile("tensor_load_to_lds %0, %1" :: "s"(g0), "s"(g1) : "memory");
}

// ------------------------------ VQ ------------------------------------------
// Codebook staged to LDS: C==32 (E0) via the Tensor Data Mover (TENSORcnt);
// smaller codebooks via global_load_async_to_lds_b128 (ASYNCcnt).
template <int C>
__global__ __launch_bounds__(256)
void vq_k(const float* __restrict__ x, const float* __restrict__ E,
          float* __restrict__ q, float* __restrict__ loss,
          int Nb, int hwsh, int Kcodes, float scale)
{
    extern __shared__ float Esh[];
    const unsigned base = __builtin_amdgcn_groupstaticsize();   // dyn-LDS start
    if (C == 32) {
        if (threadIdx.x < 32) {                 // wave 0 issues the DMA
            tdm_load_1d(E, base, (unsigned)(Kcodes * C));
            __builtin_amdgcn_s_wait_tensorcnt(0);
        }
    } else {
        const int bytes = Kcodes * C * 4;
        const char* g = (const char*)E;
        for (int off = threadIdx.x * 16; off < bytes; off += blockDim.x * 16) {
            unsigned lds_addr = base + (unsigned)off;
            unsigned long long ga = (unsigned long long)(uintptr_t)(g + off);
            asm volatile("global_load_async_to_lds_b128 %0, %1, off"
                         :: "v"(lds_addr), "v"(ga) : "memory");
        }
        asm volatile("s_wait_asynccnt 0" ::: "memory");
    }
    __syncthreads();

    const int hw = 1 << hwsh;
    const int rows = Nb << hwsh;
    const int r = blockIdx.x * blockDim.x + threadIdx.x;
    if (r >= rows) return;
    const int p = r & (hw - 1), n = r >> hwsh;

    float v[C];
    #pragma unroll
    for (int c = 0; c < C; ++c) v[c] = x[(((size_t)n * C + c) << hwsh) + p];

    int best = 0; float bestd = 3.4e38f;
    for (int k = 0; k < Kcodes; ++k) {
        const float* e = &Esh[k * C];
        float d = 0.f;
        #pragma unroll
        for (int c = 0; c < C; ++c) { float df = v[c] - e[c]; d += df * df; }
        if (d < bestd) { bestd = d; best = k; }
    }
    const float* e = &Esh[best * C];
    float sse = 0.f;
    #pragma unroll
    for (int c = 0; c < C; ++c) {
        const float qv = e[c];
        q[(((size_t)n * C + c) << hwsh) + p] = qv;
        const float df = qv - v[c]; sse += df * df;
    }
    atomicAdd(loss, sse * scale);   // loss = (1 + beta) * MSE, summed over VQs
}

__global__ void zero_one(float* p) { if (threadIdx.x == 0 && blockIdx.x == 0) *p = 0.f; }

// ---------------------------------------------------------------------------
extern "C" void kernel_launch(void* const* d_in, const int* in_sizes, int n_in,
                              void* d_out, int out_size, void* d_ws, size_t ws_size,
                              hipStream_t stream)
{
    (void)in_sizes; (void)n_in; (void)out_size; (void)ws_size;
    auto F = [&](int i) { return (const float*)d_in[i]; };

    const float* x = F(0);
    float* out = (float*)d_out;
    float* bq = out;                 // [16,32,4,4]   = 8192
    float* mq = out + 8192;          // [16,8,16,16]  = 32768
    float* tq = out + 40960;         // [16,2,64,64]  = 131072
    float* lossp = out + 172032;

    float* ws = (float*)d_ws;
    size_t off = 0;
    auto alloc = [&](size_t n) { float* p = ws + off; off += n; return p; };

    float* res0_out = alloc(16ull * 4 * 256 * 256);
    float* c1a_out  = alloc(16ull * 8 * 128 * 128);
    float* top      = alloc(16ull * 16 * 64 * 64);
    float* c2a_out  = alloc(16ull * 32 * 32 * 32);
    float* mid      = alloc(16ull * 64 * 16 * 16);
    float* c3a_out  = alloc(16ull * 128 * 8 * 8);
    float* bot      = alloc(16ull * 256 * 4 * 4);
    float* h1       = alloc(16ull * 512 * 4 * 4);
    float* res1_out = alloc(16ull * 32 * 4 * 4);
    float* t1a_out  = alloc(16ull * 16 * 8 * 8);
    float* cat1     = alloc(16ull * 72 * 16 * 16);
    float* h2       = alloc(16ull * 128 * 16 * 16);
    float* res2_out = alloc(16ull * 8 * 16 * 16);
    float* t2a_out  = alloc(16ull * 4 * 32 * 32);
    float* cat2     = alloc(16ull * 18 * 64 * 64);
    float* h3       = alloc(16ull * 32 * 64 * 64);
    float* res3_out = alloc(16ull * 2 * 64 * 64);
    // folded res-conv weights/biases
    float* wf[6]; float* bf[6];
    const int rcCo[6]  = {8, 16, 32, 64, 128, 256};
    const int rcCi[6]  = {4, 8, 16, 32, 64, 128};
    const int rcCiSh[6]= {2, 3, 4, 5, 6, 7};
    for (int i = 0; i < 6; ++i) {
        wf[i] = alloc((size_t)rcCo[i] * rcCi[i] * 16);
        bf[i] = alloc(rcCo[i]);
    }

    auto conv = [&](const float* in, const float* wgt, const float* bias, float* o,
                    int N, int Cin, int Hin, int Win, int Cout, int hwsh, int wsh,
                    int KH, int KW, int s, int pad, int co2, int Ctot, int flags) {
        dim3 g((unsigned)(((N << hwsh) + 127) / 128), (unsigned)((Cout + 15) / 16));
        conv_wmma<<<g, dim3(128), 0, stream>>>(in, wgt, bias, o, N, Cin, Hin, Win,
                                               Cout, hwsh, wsh, KH, KW, s, pad,
                                               co2, Ctot, flags);
    };

    // -------- res0 (fused) --------
    res0_fused<<<dim3(16, 16, 16), 256, 0, stream>>>(x, F(1), F(2), F(3), F(4),
                                                     F(5), F(6), res0_out, 256, 256);

    // -------- fold the 6 res-conv layers (inputs 7..30) --------
    for (int i = 0; i < 6; ++i) {
        const int base = 7 + 4 * i;
        const int tot = rcCo[i] * rcCi[i] * 16;
        fold_rc<<<dim3((tot + 255) / 256), 256, 0, stream>>>(
            F(base), F(base + 1), F(base + 2), F(base + 3), wf[i], bf[i],
            rcCo[i], rcCiSh[i]);
    }

    // -------- encoder pyramid: each res-conv == one folded WMMA GEMM --------
    conv(res0_out, wf[0], bf[0], c1a_out, 16,   4, 256, 256,   8, 14, 7, 4, 4, 2, 1, 0,   8, F_RELU);
    conv(c1a_out,  wf[1], bf[1], top,     16,   8, 128, 128,  16, 12, 6, 4, 4, 2, 1, 0,  16, F_RELU);
    conv(top,      wf[2], bf[2], c2a_out, 16,  16,  64,  64,  32, 10, 5, 4, 4, 2, 1, 0,  32, F_RELU);
    conv(c2a_out,  wf[3], bf[3], mid,     16,  32,  32,  32,  64,  8, 4, 4, 4, 2, 1, 0,  64, F_RELU);
    conv(mid,      wf[4], bf[4], c3a_out, 16,  64,  16,  16, 128,  6, 3, 4, 4, 2, 1, 0, 128, F_RELU);
    conv(c3a_out,  wf[5], bf[5], bot,     16, 128,   8,   8, 256,  4, 2, 4, 4, 2, 1, 0, 256, F_RELU);

    // -------- res1 (256 -> 512 -> 32 @ 4x4), inputs 31..36 --------
    conv(bot, F(35), F(36), res1_out, 16, 256, 4, 4,  32, 4, 2, 1, 1, 1, 0, 0,  32, 0);       // 1x1 skip
    conv(bot, F(31), F(32), h1,       16, 256, 4, 4, 512, 4, 2, 3, 3, 1, 1, 0, 512, F_RELU);  // hidden
    conv(h1,  F(33), F(34), res1_out, 16, 512, 4, 4,  32, 4, 2, 3, 3, 1, 1, 0,  32, F_ADD | F_RELU);

    zero_one<<<1, 1, 0, stream>>>(lossp);

    // -------- VQ0: E0 [512,32] (TDM path) --------
    vq_k<32><<<dim3(1), 256, 512 * 32 * 4, stream>>>(res1_out, F(65), bq, lossp,
                                                     16, 4, 512, 1.25f / 8192.f);

    // -------- decoder: t1a, t1b, concat(mid), res2 --------
    res_trans_k<<<dim3(64),  256, 0, stream>>>(bq,      F(37), F(38), F(39), F(40), t1a_out, 16, 32, 2, 2, 4, 0, 16);
    res_trans_k<<<dim3(128), 256, 0, stream>>>(t1a_out, F(41), F(42), F(43), F(44), cat1,    16, 16, 3, 3, 3, 0, 72);
    copy_ch<<<dim3(1024), 256, 0, stream>>>(mid, cat1, 16, 8, 6, 8, 72);

    conv(cat1, F(49), F(50), res2_out, 16,  72, 16, 16,   8, 8, 4, 1, 1, 1, 0, 0,   8, 0);
    conv(cat1, F(45), F(46), h2,       16,  72, 16, 16, 128, 8, 4, 3, 3, 1, 1, 0, 128, F_RELU);
    conv(h2,   F(47), F(48), res2_out, 16, 128, 16, 16,   8, 8, 4, 3, 3, 1, 1, 0,   8, F_ADD | F_RELU);

    // -------- VQ1: E1 [512,8] (async-to-LDS path) --------
    vq_k<8><<<dim3(16), 256, 512 * 8 * 4, stream>>>(res2_out, F(66), mq, lossp,
                                                    16, 8, 512, 1.25f / 32768.f);

    // -------- decoder: t2a, t2b, concat(top), res3 --------
    res_trans_k<<<dim3(256), 256, 0, stream>>>(mq,      F(51), F(52), F(53), F(54), t2a_out, 16, 8, 4, 4, 2, 0,  4);
    res_trans_k<<<dim3(512), 256, 0, stream>>>(t2a_out, F(55), F(56), F(57), F(58), cat2,    16, 4, 5, 5, 1, 0, 18);
    copy_ch<<<dim3(4096), 256, 0, stream>>>(top, cat2, 16, 12, 4, 2, 18);

    conv(cat2, F(63), F(64), res3_out, 16, 18, 64, 64,  2, 12, 6, 1, 1, 1, 0, 0,  2, 0);
    conv(cat2, F(59), F(60), h3,       16, 18, 64, 64, 32, 12, 6, 3, 3, 1, 1, 0, 32, F_RELU);
    conv(h3,   F(61), F(62), res3_out, 16, 32, 64, 64,  2, 12, 6, 3, 3, 1, 1, 0,  2, F_ADD | F_RELU);

    // -------- VQ2: E2 [512,2] (async-to-LDS path) --------
    vq_k<2><<<dim3(256), 256, 512 * 2 * 4, stream>>>(res3_out, F(67), tq, lossp,
                                                     16, 12, 512, 1.25f / 131072.f);
}
